// MultiHeadAttentionBlock_55327768707213
// MI455X (gfx1250) — matrix operand order repaired and checked by reading, not compiled, not run on hardware
//
#include <hip/hip_runtime.h>

// MI455X / gfx1250 MHA block. f16 WMMA (16x16x32, fp32 accum) for all GEMMs,
// flash-attention online softmax in fp32, TDM (tensor_load_to_lds) staging for
// the f16 output-projection A tiles.  Workspace (needs 64 MB):
//   Qp [B*H][S][64] f16, Kp [B*H][S][64] f16, Vt [B*H][64][S] f16, Ctx [B*S][1024] f16.

typedef _Float16 f16;
typedef __attribute__((ext_vector_type(16))) _Float16 v16h;
typedef __attribute__((ext_vector_type(8)))  float    v8f;
typedef __attribute__((ext_vector_type(4)))  _Float16 v4h;
typedef __attribute__((ext_vector_type(4)))  unsigned int u32x4;
typedef __attribute__((ext_vector_type(8)))  unsigned int u32x8;

union Frag16 { v16h v; uint4 q[2]; };
union G0u { u32x4 v; unsigned int u[4]; };
union G1u { u32x8 v; unsigned int u[8]; };

__device__ __forceinline__ v8f wmma_f16(const Frag16& a, const Frag16& b, v8f c) {
    return __builtin_amdgcn_wmma_f32_16x16x32_f16(false, a.v, false, b.v,
                                                  (short)0, c, false, false);
}

// ---------------------------------------------------------------------------
// TDM: async-load a 2D f16 tile [tile1 rows x tile0 elems] from a row-major
// tensor [tdim1][tdim0] (row stride = stride0 elems) into LDS at lds_addr,
// inserting pad_amount(+1) DWORDs of padding after each 2^(pad_interval+1)
// DWORDs of data (ISA D# 8.3/8.4).  Caller must s_wait_tensorcnt.
// ---------------------------------------------------------------------------
__device__ __forceinline__ void tdm_load_tile_f16(
    unsigned int lds_addr, const f16* gptr,
    unsigned int tdim0, unsigned int tdim1, unsigned int stride0,
    unsigned int tile0, unsigned int tile1,
    unsigned int pad_interval_code, unsigned int pad_amount_code)
{
    unsigned long long ga = (unsigned long long)(uintptr_t)gptr;
    G0u g0;
    g0.u[0] = 1u;                                     // count=1, user descriptor
    g0.u[1] = lds_addr;                               // LDS byte address
    g0.u[2] = (unsigned int)ga;                       // global_addr[31:0]
    g0.u[3] = (unsigned int)((ga >> 32) & 0x01FFFFFFu) | (2u << 30);  // type=2
    G1u g1;
    g1.u[0] = (1u << 16)                              // data_size = 2 bytes
            | (1u << 20)                              // pad_enable
            | (pad_interval_code << 22)
            | (pad_amount_code << 25);
    g1.u[1] = (tdim0 & 0xFFFFu) << 16;                // tensor_dim0[15:0]
    g1.u[2] = (tdim0 >> 16) | ((tdim1 & 0xFFFFu) << 16);
    g1.u[3] = (tdim1 >> 16) | (tile0 << 16);          // tile_dim0
    g1.u[4] = tile1;                                  // tile_dim1 (tile_dim2=0)
    g1.u[5] = stride0;                                // tensor_dim0_stride[31:0]
    g1.u[6] = 0u;
    g1.u[7] = 0u;
    asm volatile("tensor_load_to_lds %0, %1" :: "s"(g0.v), "s"(g1.v) : "memory");
}

// ---------------------------------------------------------------------------
// QKV projection: out = X @ W^T + bias, X [8192][1024] f32, W [1024][1024] f32.
// MODE 0: write f16 at [(b*16+h)][s][dk]     (Q and K layout)
// MODE 1: write f16 at [(b*16+h)][dk][s]     (V transposed for P*V B-frags)
// ---------------------------------------------------------------------------
template <int MODE>
__global__ __launch_bounds__(256)
void proj_kernel(const float* __restrict__ X, const float* __restrict__ W,
                 const float* __restrict__ bias, f16* __restrict__ out) {
    __shared__ alignas(16) f16 As[128 * 40];   // 128 rows x 32 k (+8 pad)
    __shared__ alignas(16) f16 Bs[128 * 40];   // 128 cols x 32 k (+8 pad)

    const int tid  = threadIdx.x;
    const int lane = tid & 31;
    const int wave = tid >> 5;
    const int wrow = wave >> 2;        // 0..1  (64 rows each)
    const int wcol = wave & 3;         // 0..3  (32 cols each)
    const int m    = lane & 15;
    const int hi   = lane >> 4;
    const int cbase = blockIdx.x * 128;
    const int rbase = blockIdx.y * 128;

    const v8f zero = {0.f,0.f,0.f,0.f,0.f,0.f,0.f,0.f};
    v8f acc[4][2];
    #pragma unroll
    for (int mt = 0; mt < 4; ++mt)
        #pragma unroll
        for (int nt = 0; nt < 2; ++nt) acc[mt][nt] = zero;

    for (int kb = 0; kb < 1024; kb += 32) {
        // Stage fp32 -> f16 tiles into LDS.
        #pragma unroll
        for (int i = 0; i < 4; ++i) {
            int id = tid + 256 * i;          // 1024 float4 per operand
            int r  = id >> 3;                // 0..127
            int kq = (id & 7) << 2;          // 0..28
            float4 xa = *(const float4*)(X + (size_t)(rbase + r) * 1024 + kb + kq);
            float4 xb = *(const float4*)(W + (size_t)(cbase + r) * 1024 + kb + kq);
            v4h ha; ha.x = (f16)xa.x; ha.y = (f16)xa.y; ha.z = (f16)xa.z; ha.w = (f16)xa.w;
            v4h hb; hb.x = (f16)xb.x; hb.y = (f16)xb.y; hb.z = (f16)xb.z; hb.w = (f16)xb.w;
            *(v4h*)&As[r * 40 + kq] = ha;
            *(v4h*)&Bs[r * 40 + kq] = hb;
        }
        __syncthreads();

        Frag16 a[4], bf[2];
        #pragma unroll
        for (int mt = 0; mt < 4; ++mt) {
            const f16* p = &As[(wrow * 64 + mt * 16 + m) * 40];
            a[mt].q[0] = *(const uint4*)(p + hi * 8);         // K = (hi?8:0)+0..7
            a[mt].q[1] = *(const uint4*)(p + 16 + hi * 8);    // K = 16+(hi?8:0)+0..7
        }
        #pragma unroll
        for (int nt = 0; nt < 2; ++nt) {
            const f16* p = &Bs[(wcol * 32 + nt * 16 + m) * 40];
            bf[nt].q[0] = *(const uint4*)(p + hi * 8);        // K = (hi?8:0)+0..7
            bf[nt].q[1] = *(const uint4*)(p + hi * 8 + 16);   // K = 16+(hi?8:0)+0..7
        }
        #pragma unroll
        for (int mt = 0; mt < 4; ++mt)
            #pragma unroll
            for (int nt = 0; nt < 2; ++nt)
                acc[mt][nt] = wmma_f16(a[mt], bf[nt], acc[mt][nt]);
        __syncthreads();
    }

    // Epilogue: bias + f16 store in per-head layout.
    #pragma unroll
    for (int nt = 0; nt < 2; ++nt) {
        const int col = cbase + wcol * 32 + nt * 16 + m;
        const float bv = bias[col];
        const int h  = col >> 6;
        const int dk = col & 63;
        #pragma unroll
        for (int mt = 0; mt < 4; ++mt) {
            #pragma unroll
            for (int r = 0; r < 8; ++r) {
                int rowg = rbase + wrow * 64 + mt * 16 + r + 8 * hi;
                int b = rowg >> 11, s = rowg & 2047;
                float val = acc[mt][nt][r] + bv;
                size_t base = (size_t)(b * 16 + h) * 131072;
                if (MODE == 0) out[base + (size_t)s * 64 + dk] = (f16)val;
                else           out[base + (size_t)dk * 2048 + s] = (f16)val;
            }
        }
    }
}

// ---------------------------------------------------------------------------
// Flash attention: each wave owns 16 query rows; loop key chunks of 64.
// Scores fp32-accumulated from f16 WMMA; online softmax in fp32.
// ---------------------------------------------------------------------------
__global__ __launch_bounds__(256)
void attn_kernel(const f16* __restrict__ Qp, const f16* __restrict__ Kp,
                 const f16* __restrict__ Vt, const int* __restrict__ mask,
                 f16* __restrict__ Ctx) {
    __shared__ alignas(16) f16 Ps[8][16 * 72];  // per-wave 16x64 P tile (+8 pad)

    const int tid  = threadIdx.x;
    const int lane = tid & 31;
    const int wave = tid >> 5;
    const int m    = lane & 15;
    const int hi   = lane >> 4;
    const int bh   = blockIdx.y;
    const int b    = bh >> 4;
    const int h    = bh & 15;
    const size_t hoff = (size_t)bh * 131072;       // 2048*64
    const int qrow = blockIdx.x * 128 + wave * 16;

    // Q A-fragments for this wave's 16 rows (d_k = 64 -> two K=32 frags).
    Frag16 aq[2];
    #pragma unroll
    for (int kk = 0; kk < 2; ++kk) {
        const f16* p = Qp + hoff + (size_t)(qrow + m) * 64 + kk * 32 + hi * 8;
        aq[kk].q[0] = *(const uint4*)p;
        aq[kk].q[1] = *(const uint4*)(p + 16);
    }

    const v8f zero = {0.f,0.f,0.f,0.f,0.f,0.f,0.f,0.f};
    v8f acc[4];
    #pragma unroll
    for (int nt = 0; nt < 4; ++nt) acc[nt] = zero;
    float mrun[8], lrun[8];
    #pragma unroll
    for (int r = 0; r < 8; ++r) { mrun[r] = -3.0e38f; lrun[r] = 0.f; }

    f16* ps = &Ps[wave][0];
    const int* mrow = mask + b * 2048;

    for (int kc = 0; kc < 2048; kc += 64) {
        // --- scores S = Q K^T for 64 keys ---
        v8f sc[4];
        #pragma unroll
        for (int nt = 0; nt < 4; ++nt) sc[nt] = zero;
        #pragma unroll
        for (int nt = 0; nt < 4; ++nt) {
            const f16* kp = Kp + hoff + (size_t)(kc + nt * 16 + m) * 64 + hi * 8;
            #pragma unroll
            for (int kk = 0; kk < 2; ++kk) {
                Frag16 bk;
                bk.q[0] = *(const uint4*)(kp + kk * 32);
                bk.q[1] = *(const uint4*)(kp + kk * 32 + 16);
                sc[nt] = wmma_f16(aq[kk], bk, sc[nt]);
            }
        }
        // --- scale + mask + chunk row max (N is across lanes 0..15 / 16..31) ---
        float cm[8];
        #pragma unroll
        for (int r = 0; r < 8; ++r) cm[r] = -3.0e38f;
        #pragma unroll
        for (int nt = 0; nt < 4; ++nt) {
            int mv = mrow[kc + nt * 16 + m];
            #pragma unroll
            for (int r = 0; r < 8; ++r) {
                float v = sc[nt][r] * 0.125f;          // 1/sqrt(64)
                v = (mv == 0) ? -1.0e9f : v;
                sc[nt][r] = v;
                cm[r] = fmaxf(cm[r], v);
            }
        }
        #pragma unroll
        for (int r = 0; r < 8; ++r)
            #pragma unroll
            for (int off = 8; off >= 1; off >>= 1)
                cm[r] = fmaxf(cm[r], __shfl_xor(cm[r], off, 32));
        // --- online softmax update ---
        float alpha[8];
        #pragma unroll
        for (int r = 0; r < 8; ++r) {
            float nm = fmaxf(mrun[r], cm[r]);
            alpha[r] = __expf(mrun[r] - nm);
            mrun[r] = nm;
        }
        float cs[8];
        #pragma unroll
        for (int r = 0; r < 8; ++r) cs[r] = 0.f;
        #pragma unroll
        for (int nt = 0; nt < 4; ++nt)
            #pragma unroll
            for (int r = 0; r < 8; ++r) {
                float p = __expf(sc[nt][r] - mrun[r]);
                sc[nt][r] = p;
                cs[r] += p;
            }
        #pragma unroll
        for (int r = 0; r < 8; ++r) {
            #pragma unroll
            for (int off = 8; off >= 1; off >>= 1)
                cs[r] += __shfl_xor(cs[r], off, 32);
            lrun[r] = lrun[r] * alpha[r] + cs[r];
        }
        #pragma unroll
        for (int nt = 0; nt < 4; ++nt)
            #pragma unroll
            for (int r = 0; r < 8; ++r) acc[nt][r] *= alpha[r];

        // --- P: C-layout -> A-layout via per-wave LDS bounce (f16) ---
        #pragma unroll
        for (int nt = 0; nt < 4; ++nt)
            #pragma unroll
            for (int r = 0; r < 8; ++r)
                ps[(r + 8 * hi) * 72 + nt * 16 + m] = (f16)sc[nt][r];
        Frag16 ap[2];
        #pragma unroll
        for (int kk = 0; kk < 2; ++kk) {
            const f16* p = ps + m * 72 + kk * 32 + hi * 8;
            ap[kk].q[0] = *(const uint4*)p;
            ap[kk].q[1] = *(const uint4*)(p + 16);
        }
        // --- O += P V  (Vt is [dk][s] so B-frags are contiguous) ---
        #pragma unroll
        for (int nt = 0; nt < 4; ++nt) {
            const f16* vp = Vt + hoff + (size_t)(nt * 16 + m) * 2048 + kc + hi * 8;
            #pragma unroll
            for (int kk = 0; kk < 2; ++kk) {
                Frag16 bv;
                bv.q[0] = *(const uint4*)(vp + kk * 32);
                bv.q[1] = *(const uint4*)(vp + kk * 32 + 16);
                acc[nt] = wmma_f16(ap[kk], bv, acc[nt]);
            }
        }
    }

    // Normalize and write context in [b][s][h*64+dk] (row-major 8192x1024 f16).
    float inv[8];
    #pragma unroll
    for (int r = 0; r < 8; ++r) inv[r] = 1.0f / lrun[r];
    #pragma unroll
    for (int nt = 0; nt < 4; ++nt) {
        int dk = nt * 16 + m;
        #pragma unroll
        for (int r = 0; r < 8; ++r) {
            int s = qrow + r + 8 * hi;
            float o = acc[nt][r] * inv[r];
            Ctx[(size_t)(b * 2048 + s) * 1024 + h * 64 + dk] = (f16)o;
        }
    }
}

// ---------------------------------------------------------------------------
// Output projection: out = Ctx(f16) @ W^T + bias -> fp32.
// A tile (already f16) staged via the Tensor Data Mover with LDS padding
// (row = 16 data DWORDs + 4 pad DWORDs -> 40-half stride, matching the
// WMMA fragment reader).  W tile converted fp32->f16 by the threads.
// ---------------------------------------------------------------------------
__global__ __launch_bounds__(256)
void outproj_kernel(const f16* __restrict__ Xh, const float* __restrict__ W,
                    const float* __restrict__ bias, float* __restrict__ out) {
    __shared__ alignas(16) f16 As[128 * 40];
    __shared__ alignas(16) f16 Bs[128 * 40];

    const int tid  = threadIdx.x;
    const int lane = tid & 31;
    const int wave = tid >> 5;
    const int wrow = wave >> 2;
    const int wcol = wave & 3;
    const int m    = lane & 15;
    const int hi   = lane >> 4;
    const int cbase = blockIdx.x * 128;
    const int rbase = blockIdx.y * 128;

    const unsigned int lds_as = (unsigned int)(uintptr_t)&As[0];

    const v8f zero = {0.f,0.f,0.f,0.f,0.f,0.f,0.f,0.f};
    v8f acc[4][2];
    #pragma unroll
    for (int mt = 0; mt < 4; ++mt)
        #pragma unroll
        for (int nt = 0; nt < 2; ++nt) acc[mt][nt] = zero;

    for (int kb = 0; kb < 1024; kb += 32) {
        // A tile via TDM: 128 rows x 32 halves from Ctx[rbase..][kb..].
        // pad_interval code 3 = 16 DWORDs of data (one 32-half row),
        // pad_amount  code 3 = 4 DWORDs pad  -> 40-half LDS row stride.
        if (wave == 0) {
            tdm_load_tile_f16(lds_as, Xh + (size_t)rbase * 1024 + kb,
                              /*tdim0=*/1024, /*tdim1=*/8192, /*stride0=*/1024,
                              /*tile0=*/32, /*tile1=*/128,
                              /*pad_interval=*/3u, /*pad_amount=*/3u);
        }
        // B fp32 -> f16 convert (overlaps with the TDM transfer).
        #pragma unroll
        for (int i = 0; i < 4; ++i) {
            int id = tid + 256 * i;
            int r  = id >> 3;
            int kq = (id & 7) << 2;
            float4 xb = *(const float4*)(W + (size_t)(cbase + r) * 1024 + kb + kq);
            v4h hb; hb.x = (f16)xb.x; hb.y = (f16)xb.y; hb.z = (f16)xb.z; hb.w = (f16)xb.w;
            *(v4h*)&Bs[r * 40 + kq] = hb;
        }
        if (wave == 0) __builtin_amdgcn_s_wait_tensorcnt(0);
        __syncthreads();

        Frag16 a[4], bf[2];
        #pragma unroll
        for (int mt = 0; mt < 4; ++mt) {
            const f16* p = &As[(wrow * 64 + mt * 16 + m) * 40];
            a[mt].q[0] = *(const uint4*)(p + hi * 8);
            a[mt].q[1] = *(const uint4*)(p + 16 + hi * 8);
        }
        #pragma unroll
        for (int nt = 0; nt < 2; ++nt) {
            const f16* p = &Bs[(wcol * 32 + nt * 16 + m) * 40];
            bf[nt].q[0] = *(const uint4*)(p + hi * 8);
            bf[nt].q[1] = *(const uint4*)(p + hi * 8 + 16);
        }
        #pragma unroll
        for (int mt = 0; mt < 4; ++mt)
            #pragma unroll
            for (int nt = 0; nt < 2; ++nt)
                acc[mt][nt] = wmma_f16(a[mt], bf[nt], acc[mt][nt]);
        __syncthreads();
    }

    #pragma unroll
    for (int nt = 0; nt < 2; ++nt) {
        const int col = cbase + wcol * 32 + nt * 16 + m;
        const float bv = bias[col];
        #pragma unroll
        for (int mt = 0; mt < 4; ++mt) {
            #pragma unroll
            for (int r = 0; r < 8; ++r) {
                int rowg = rbase + wrow * 64 + mt * 16 + r + 8 * hi;
                out[(size_t)rowg * 1024 + col] = acc[mt][nt][r] + bv;
            }
        }
    }
}

// ---------------------------------------------------------------------------
extern "C" void kernel_launch(void* const* d_in, const int* in_sizes, int n_in,
                              void* d_out, int out_size, void* d_ws, size_t ws_size,
                              hipStream_t stream) {
    (void)in_sizes; (void)n_in; (void)out_size; (void)ws_size;
    const float* q   = (const float*)d_in[0];
    const float* k   = (const float*)d_in[1];
    const float* v   = (const float*)d_in[2];
    const int*   msk = (const int*)  d_in[3];
    const float* w_q = (const float*)d_in[4];
    const float* b_q = (const float*)d_in[5];
    const float* w_k = (const float*)d_in[6];
    const float* b_k = (const float*)d_in[7];
    const float* w_v = (const float*)d_in[8];
    const float* b_v = (const float*)d_in[9];
    const float* w_o = (const float*)d_in[10];
    const float* b_o = (const float*)d_in[11];

    f16* ws = (f16*)d_ws;
    const size_t NE = (size_t)8192 * 1024;   // elements per f16 tensor
    f16* Qp  = ws;
    f16* Kp  = ws + NE;
    f16* Vt  = ws + 2 * NE;
    f16* Ctx = ws + 3 * NE;                  // 64 MB workspace total

    dim3 blk(256);
    dim3 pg(8, 64);    // cols/128, rows/128
    dim3 ag(16, 64);   // S/128 query tiles, B*H

    proj_kernel<0><<<pg, blk, 0, stream>>>(q, w_q, b_q, Qp);
    proj_kernel<0><<<pg, blk, 0, stream>>>(k, w_k, b_k, Kp);
    proj_kernel<1><<<pg, blk, 0, stream>>>(v, w_v, b_v, Vt);
    attn_kernel<<<ag, blk, 0, stream>>>(Qp, Kp, Vt, msk, Ctx);
    outproj_kernel<<<pg, blk, 0, stream>>>(Ctx, w_o, b_o, (float*)d_out);
}
